// DissipationNetwork_35029753266887
// MI455X (gfx1250) — matrix operand order repaired and checked by reading, not compiled, not run on hardware
//
#include <hip/hip_runtime.h>

typedef __attribute__((ext_vector_type(16))) _Float16 v16h;
typedef __attribute__((ext_vector_type(8)))  _Float16 v8h;
typedef __attribute__((ext_vector_type(8)))  float    v8f;

#define DVC __device__ __forceinline__

// ---------------- packed weight layout (f16 element offsets in d_ws) --------
// Biases are folded into a padding column: col 16 for Kp=32 mats, col 50 for
// Kp=64 mats.  A operands carry 1.0 in that K slot.
constexpr int OFF_xin   = 0;      // 64x32, bias col 16
constexpr int OFF_xl1   = 2048;   // 64x32, bias col 16
constexpr int OFF_clin  = 4096;   // 64x32, no bias
constexpr int OFF_cl1   = 6144;   // 64x32, no bias
constexpr int OFF_cl2   = 8192;   // 64x32, no bias
constexpr int OFF_xp1   = 10240;  // 64x64, bias col 50
constexpr int OFF_xp2   = 14336;  // 64x64, bias col 50
constexpr int OFF_xl2   = 18432;  // 64x64, bias col 50
constexpr int OFF_xl3   = 22528;  // 64x64, bias col 50
constexpr int OFF_cp1   = 26624;  // 64x64, no bias
constexpr int OFF_cp2   = 30720;  // 64x64, no bias
constexpr int OFF_cp1m  = 34816;  // 64x64, bias col 50
constexpr int OFF_cp2m  = 38912;  // 64x64, bias col 50
constexpr int OFF_cpom  = 43008;  // 64x64, bias col 50
constexpr int OFF_xlo   = 47104;  // 16x64, bias col 50
constexpr int OFF_cpo   = 48128;  // 16x64, no bias
constexpr int OFF_cl1m  = 49152;  // 16x64, bias col 50
constexpr int OFF_cl2m  = 50176;  // 16x64, bias col 50
constexpr int OFF_clom  = 51200;  // 16x64, bias col 50
constexpr int OFF_clo   = 52224;  // 16x32, no bias
constexpr int OFF_clinm = 52736;  // 16x32, bias col 16
constexpr int WELEMS = 53248;
constexpr int WBYTES = WELEMS * 2;   // 106496, multiple of 16

// ---------------- WMMA helpers ----------------------------------------------
DVC v8f wm(v16h a, v16h b, v8f c) {
    return __builtin_amdgcn_wmma_f32_16x16x32_f16(false, a, false, b,
                                                  (short)0, c, false, false);
}
DVC v8f wm0(v16h a, v16h b) {          // zero C -> inline-0 SRC2, no movs
    v8f z = {};
    return __builtin_amdgcn_wmma_f32_16x16x32_f16(false, a, false, b,
                                                  (short)0, z, false, false);
}

// B fragment: 32x16 f16 (KxN). lane n (0-15): col n, K=0..7,16..23; lane n+16: K=8..15,24..31.
// Weight stored row-major [Np][Kp]; each lane reads two 8-f16 chunks (16B each).
DVC v16h ldsB(const _Float16* W, int Kp, int nt, int kt, int lane) {
    int n  = nt * 16 + (lane & 15);
    int k0 = kt * 32 + ((lane >> 4) << 3);
    const _Float16* p = W + n * Kp + k0;
    v8h lo = *(const v8h*)(p);
    v8h hi = *(const v8h*)(p + 16);
    v16h r;
#pragma unroll
    for (int i = 0; i < 8; ++i) { r[i] = lo[i]; r[i + 8] = hi[i]; }
    return r;
}

// A fragment from 16x64 f16 row-major scratch.
// lanes 0-15: M=lane, K=k0+0..7 / k0+16..23 ; lanes 16-31: K=k0+8..15 / k0+24..31
DVC v16h ldsA(const _Float16* s, int kt, int lane) {
    int m   = lane & 15;
    int off = kt * 32 + ((lane >> 4) ? 8 : 0);
    const _Float16* p = s + m * 64;
    v8h lo = *(const v8h*)(p + off);
    v8h hi = *(const v8h*)(p + off + 16);
    v16h r;
#pragma unroll
    for (int i = 0; i < 8; ++i) { r[i] = lo[i]; r[i + 8] = hi[i]; }
    return r;
}

// store f32 C/D fragment as f16 into 16x64 scratch.
// MODE 0: plain; MODE 1: also zero cols 16..31; MODE 2: force 1.0 at col 50
template <int MODE>
DVC void stD(_Float16* s, int nt, v8f d, int lane) {
    int col = nt * 16 + (lane & 15);
    int rb  = (lane >> 4) * 8;
    bool one = (MODE == 2) && (col == 50);
#pragma unroll
    for (int g = 0; g < 8; ++g) {
        _Float16 h = (_Float16)d[g];
        if (MODE == 2) h = one ? (_Float16)1.f : h;
        s[(rb + g) * 64 + col] = h;
        if (MODE == 1) s[(rb + g) * 64 + col + 16] = (_Float16)0.f;
    }
}

DVC v8f spv(v8f x) {  // softplus, numerically stable
    v8f r;
#pragma unroll
    for (int g = 0; g < 8; ++g) {
        float xi = x[g];
        float e  = __expf(-fabsf(xi));
        r[g] = fmaxf(xi, 0.f) + __logf(1.f + e);
    }
    return r;
}

// ---------------- fused stages ----------------------------------------------
DVC void midStage(v8f* Xx, v8f* Xs, v8f x0s, _Float16* sH,
                  const _Float16* sW, int lane,
                  int off_xl, int off_cpm, int off_cp,
                  int off_clm, int off_cl, v16h* axOut) {
#pragma unroll
    for (int nt = 0; nt < 4; ++nt) stD<2>(sH, nt, Xx[nt], lane);  // x (+ones col)
    v16h ax0 = ldsA(sH, 0, lane), ax1 = ldsA(sH, 1, lane);
    axOut[0] = ax0; axOut[1] = ax1;

    v8f T[4];
#pragma unroll
    for (int nt = 0; nt < 4; ++nt) {                 // sp(x@Wcpm.T + b) gate
        v8f g = wm0(ax0, ldsB(sW + off_cpm, 64, nt, 0, lane));
        g = wm(ax1, ldsB(sW + off_cpm, 64, nt, 1, lane), g);
        T[nt] = Xs[nt] * spv(g);
    }
    v8f u = wm0(ax0, ldsB(sW + off_clm, 64, 0, 0, lane));  // x@Wclm.T + b
    u = wm(ax1, ldsB(sW + off_clm, 64, 0, 1, lane), u);
    u = u * x0s;

    v8f A[4];
#pragma unroll
    for (int nt = 0; nt < 4; ++nt) {                 // x@Wxl.T + b
        A[nt] = wm0(ax0, ldsB(sW + off_xl, 64, nt, 0, lane));
        A[nt] = wm(ax1, ldsB(sW + off_xl, 64, nt, 1, lane), A[nt]);
    }
#pragma unroll
    for (int nt = 0; nt < 4; ++nt) stD<0>(sH, nt, T[nt], lane);
    v16h at0 = ldsA(sH, 0, lane), at1 = ldsA(sH, 1, lane);
#pragma unroll
    for (int nt = 0; nt < 4; ++nt) {                 // += T @ Wcp.T
        A[nt] = wm(at0, ldsB(sW + off_cp, 64, nt, 0, lane), A[nt]);
        A[nt] = wm(at1, ldsB(sW + off_cp, 64, nt, 1, lane), A[nt]);
    }
    stD<1>(sH, 0, u, lane);
    v16h au = ldsA(sH, 0, lane);
#pragma unroll
    for (int nt = 0; nt < 4; ++nt)                   // += u @ Wcl.T  (Kp=32)
        A[nt] = wm(au, ldsB(sW + off_cl, 32, nt, 0, lane), A[nt]);
#pragma unroll
    for (int nt = 0; nt < 4; ++nt) Xs[nt] = spv(A[nt]);
}

DVC void xpStage(v8f* Xx, const v16h* ax, const _Float16* sW, int lane, int off) {
#pragma unroll
    for (int nt = 0; nt < 4; ++nt) {
        v8f a = wm0(ax[0], ldsB(sW + off, 64, nt, 0, lane));
        a = wm(ax[1], ldsB(sW + off, 64, nt, 1, lane), a);
        Xx[nt] = spv(a);
    }
}

// ---------------- main kernel ------------------------------------------------
__global__ __launch_bounds__(256)
void dissip_main(const float* __restrict__ inp, float* __restrict__ out,
                 const _Float16* __restrict__ gW, int NT) {
    __shared__ uint4 sWq[WBYTES / 16];       // 106496 B packed f16 weights
    __shared__ uint4 sScrQ[8 * 128];         // 2 KB scratch per wave

    const int tid  = threadIdx.x;
    const int lane = tid & 31;
    const int wave = tid >> 5;

    {   // cooperative weight stage into LDS (L2-resident source)
        const uint4* gq = (const uint4*)gW;
        for (int i = tid; i < WBYTES / 16; i += 256) sWq[i] = gq[i];
    }
    __syncthreads();

    const _Float16* sW = (const _Float16*)sWq;
    _Float16* sH = (_Float16*)(sScrQ + wave * 128);
    float*    sF = (float*)sH;

    const int m  = lane & 15;
    const int hf = lane >> 4;
    const int stride = gridDim.x * 8;

    for (int tile = blockIdx.x * 8 + wave; tile < NT; tile += stride) {
        if (tile + stride < NT)
            __builtin_prefetch(inp + (long)(tile + stride) * 512, 0, 1);

        // ---- stage 0: stream 16 rows (32 f32 each) into scratch ------------
        {
            const float4* ip = (const float4*)(inp + ((long)tile * 16 + m) * 32 + hf * 16);
            float4* sf4 = (float4*)(sF + m * 32 + hf * 16);
            sf4[0] = ip[0]; sf4[1] = ip[1]; sf4[2] = ip[2]; sf4[3] = ip[3];
        }
        v16h ax0;                          // x0 as A: K=0..15 data, K=16 -> 1.0
        {
            const float* p = sF + m * 32 + (hf ? 8 : 0);
#pragma unroll
            for (int i = 0; i < 8; ++i) { ax0[i] = (_Float16)p[i]; ax0[i + 8] = (_Float16)0.f; }
            if (!hf) ax0[8] = (_Float16)1.f;     // bias pickup column
        }
        v8f x0s;                           // x0s in C/D layout
        {
            int rb = hf * 8;
#pragma unroll
            for (int g = 0; g < 8; ++g) x0s[g] = sF[(rb + g) * 32 + 16 + m];
        }

        // ---- stage 1: xs = sp(x0@Wxl1.T+b + (x0s*(x0@Wclinm.T+b))@Wclin.T) -
        v8f t = wm0(ax0, ldsB(sW + OFF_clinm, 32, 0, 0, lane));
        t = t * x0s;
        stD<1>(sH, 0, t, lane);
        v16h at = ldsA(sH, 0, lane);
        v8f Xs[4], Xx[4];
#pragma unroll
        for (int nt = 0; nt < 4; ++nt) {
            v8f a = wm0(ax0, ldsB(sW + OFF_xl1, 32, nt, 0, lane));
            a = wm(at,  ldsB(sW + OFF_clin, 32, nt, 0, lane), a);
            Xs[nt] = spv(a);
        }

        // ---- stage 2: x = sp(x0@Wxin.T + b) --------------------------------
#pragma unroll
        for (int nt = 0; nt < 4; ++nt)
            Xx[nt] = spv(wm0(ax0, ldsB(sW + OFF_xin, 32, nt, 0, lane)));

        v16h ax[2];
        // ---- stage 3 + 4 ---------------------------------------------------
        midStage(Xx, Xs, x0s, sH, sW, lane,
                 OFF_xl2, OFF_cp1m, OFF_cp1, OFF_cl1m, OFF_cl1, ax);
        xpStage(Xx, ax, sW, lane, OFF_xp1);
        // ---- stage 5 + 6 ---------------------------------------------------
        midStage(Xx, Xs, x0s, sH, sW, lane,
                 OFF_xl3, OFF_cp2m, OFF_cp2, OFF_cl2m, OFF_cl2, ax);
        xpStage(Xx, ax, sW, lane, OFF_xp2);

        // ---- stage 7: scalar head -----------------------------------------
#pragma unroll
        for (int nt = 0; nt < 4; ++nt) stD<2>(sH, nt, Xx[nt], lane);
        v16h fx0 = ldsA(sH, 0, lane), fx1 = ldsA(sH, 1, lane);
        v8f T[4];
#pragma unroll
        for (int nt = 0; nt < 4; ++nt) {
            v8f g = wm0(fx0, ldsB(sW + OFF_cpom, 64, nt, 0, lane));
            g = wm(fx1, ldsB(sW + OFF_cpom, 64, nt, 1, lane), g);
            T[nt] = Xs[nt] * spv(g);
        }
        v8f u = wm0(fx0, ldsB(sW + OFF_clom, 64, 0, 0, lane));
        u = wm(fx1, ldsB(sW + OFF_clom, 64, 0, 1, lane), u);
        u = u * x0s;

        v8f acc = wm0(fx0, ldsB(sW + OFF_xlo, 64, 0, 0, lane));
        acc = wm(fx1, ldsB(sW + OFF_xlo, 64, 0, 1, lane), acc);
#pragma unroll
        for (int nt = 0; nt < 4; ++nt) stD<0>(sH, nt, T[nt], lane);
        v16h at0 = ldsA(sH, 0, lane), at1 = ldsA(sH, 1, lane);
        acc = wm(at0, ldsB(sW + OFF_cpo, 64, 0, 0, lane), acc);
        acc = wm(at1, ldsB(sW + OFF_cpo, 64, 0, 1, lane), acc);
        stD<1>(sH, 0, u, lane);
        v16h au = ldsA(sH, 0, lane);
        acc = wm(au, ldsB(sW + OFF_clo, 32, 0, 0, lane), acc);

        v8f res = spv(acc);
        if (m == 0) {                      // column 0 holds the scalar output
            float* op = out + (long)tile * 16 + hf * 8;
#pragma unroll
            for (int g = 0; g < 8; ++g) op[g] = res[g];
        }
    }
}

// ---------------- weight packing kernel --------------------------------------
__global__ void pack_w(const float* __restrict__ src, const float* __restrict__ bias,
                       _Float16* __restrict__ dst,
                       int rows, int cols, int Np, int Kp, int bcol) {
    int i = blockIdx.x * blockDim.x + threadIdx.x;
    if (i >= Np * Kp) return;
    int n = i / Kp, k = i % Kp;
    float v = 0.f;
    if (n < rows) {
        if (k < cols)            v = src[n * cols + k];
        else if (k == bcol)      v = bias[n];
    }
    dst[i] = (_Float16)v;
}

// ---------------- host launcher ---------------------------------------------
extern "C" void kernel_launch(void* const* d_in, const int* in_sizes, int n_in,
                              void* d_out, int out_size, void* d_ws, size_t ws_size,
                              hipStream_t stream) {
    _Float16* wsW = (_Float16*)d_ws;

    struct WRec { int wi, bi, off, rows, cols, Np, Kp, bcol; };
    static const WRec WT[21] = {
        { 1,  2, OFF_xin,   50, 16, 64, 32, 16},
        { 7,  8, OFF_xl1,   50, 16, 64, 32, 16},
        {24, -1, OFF_clin,  50, 16, 64, 32, -1},
        {25, -1, OFF_cl1,   50, 16, 64, 32, -1},
        {26, -1, OFF_cl2,   50, 16, 64, 32, -1},
        { 3,  4, OFF_xp1,   50, 50, 64, 64, 50},
        { 5,  6, OFF_xp2,   50, 50, 64, 64, 50},
        { 9, 10, OFF_xl2,   50, 50, 64, 64, 50},
        {11, 12, OFF_xl3,   50, 50, 64, 64, 50},
        {15, -1, OFF_cp1,   50, 50, 64, 64, -1},
        {16, -1, OFF_cp2,   50, 50, 64, 64, -1},
        {18, 19, OFF_cp1m,  50, 50, 64, 64, 50},
        {20, 21, OFF_cp2m,  50, 50, 64, 64, 50},
        {22, 23, OFF_cpom,  50, 50, 64, 64, 50},
        {13, 14, OFF_xlo,    1, 50, 16, 64, 50},
        {17, -1, OFF_cpo,    1, 50, 16, 64, -1},
        {30, 31, OFF_cl1m,  16, 50, 16, 64, 50},
        {32, 33, OFF_cl2m,  16, 50, 16, 64, 50},
        {34, 35, OFF_clom,  16, 50, 16, 64, 50},
        {27, -1, OFF_clo,    1, 16, 16, 32, -1},
        {28, 29, OFF_clinm, 16, 16, 16, 32, 16},
    };
    for (int i = 0; i < 21; ++i) {
        int tot = WT[i].Np * WT[i].Kp;
        const float* b = (WT[i].bi >= 0) ? (const float*)d_in[WT[i].bi] : nullptr;
        pack_w<<<(tot + 255) / 256, 256, 0, stream>>>(
            (const float*)d_in[WT[i].wi], b, wsW + WT[i].off,
            WT[i].rows, WT[i].cols, WT[i].Np, WT[i].Kp, WT[i].bcol);
    }

    int Btot = in_sizes[0] / 32;      // batch rows
    int NT   = Btot / 16;             // 16-row tiles
    int grid = (NT + 7) / 8;
    if (grid > 1024) grid = 1024;
    if (grid < 1) grid = 1;
    dissip_main<<<grid, 256, 0, stream>>>((const float*)d_in[0], (float*)d_out,
                                          wsW, NT);
}
